// BahdanauAttention_21492016349219
// MI455X (gfx1250) — compile-verified, matrix operand order checked by reading, not run on hardware
//
#include <hip/hip_runtime.h>
#include <hip/hip_bf16.h>
#include <math.h>

// ---------------------------------------------------------------------------
// Bahdanau attention, fused for MI455X (gfx1250, wave32, WMMA bf16 path).
// B=64, T=2048, D=256, U=256.
// ---------------------------------------------------------------------------

#define B_SZ 64
#define T_SZ 2048
#define D_SZ 256
#define U_SZ 256

typedef __attribute__((ext_vector_type(16))) __bf16 v16bf;
typedef __attribute__((ext_vector_type(8)))  float  v8f;

// Hardware V_TANH_F32 on gfx1250; clamped exp fallback otherwise.
__device__ inline float fast_tanh(float x) {
#if __has_builtin(__builtin_amdgcn_tanhf)
    return __builtin_amdgcn_tanhf(x);
#else
    x = fminf(fmaxf(x, -9.0f), 9.0f);
    const float e = __expf(2.0f * x);
    return (e - 1.0f) / (e + 1.0f);
#endif
}

// ---------------------------------------------------------------------------
// Kernel 1a: combo[b,u] = query[b,:] @ W1[:,u] + W1_b[u] + W2_b[u]
// ---------------------------------------------------------------------------
__global__ void combo_kernel(const float* __restrict__ query,
                             const float* __restrict__ W1,
                             const float* __restrict__ W1b,
                             const float* __restrict__ W2b,
                             float* __restrict__ combo) {
    __shared__ float q[D_SZ];
    const int b = blockIdx.x;
    const int u = threadIdx.x;           // 256 threads
    q[u] = query[b * D_SZ + u];
    __syncthreads();
    float s = W1b[u] + W2b[u];
#pragma unroll 8
    for (int d = 0; d < D_SZ; ++d)
        s += q[d] * W1[d * U_SZ + u];    // coalesced over u
    combo[b * U_SZ + u] = s;
}

// ---------------------------------------------------------------------------
// Kernel 1b: pack W2 [D,U] fp32 -> bf16 WMMA B-fragments.
// Fragment (nt, ks) covers N = nt*16..+15, K = ks*32..+31.
// 16-bit operand layout: lanes 0-15 hold K in {0..7, 16..23},
// lanes 16-31 hold K in {8..15, 24..31}; column N = lane & 15.
// Stored as [frag][lane][16 bf16] -> one 32B chunk per lane.
// ---------------------------------------------------------------------------
__global__ void pack_w2_kernel(const float* __restrict__ W2,
                               __bf16* __restrict__ pack) {
    const int frag = blockIdx.x;         // 0..127 : frag = nt*8 + ks
    const int nt   = frag >> 3;
    const int ks   = frag & 7;
    const int lane = threadIdx.x;        // 32 threads
    const int n    = nt * 16 + (lane & 15);
    const int k0   = ks * 32 + ((lane < 16) ? 0 : 8);
    __bf16* dst = pack + ((size_t)frag * 32 + lane) * 16;
#pragma unroll
    for (int j = 0; j < 16; ++j) {
        const int k = (j < 8) ? (k0 + j) : (k0 + 8 + j);   // {k0..k0+7, k0+16..k0+23}
        dst[j] = (__bf16)W2[k * U_SZ + n];
    }
}

// ---------------------------------------------------------------------------
// Helper: build one 16x32 bf16 A-fragment for this lane from fp32 values.
// p points at values row + k0 (k0 already includes the lane's half offset).
// ---------------------------------------------------------------------------
__device__ inline v16bf load_a_frag(const float* p) {
    union { v16bf v; __bf16 e[16]; } a;
    const float4 f0 = *(const float4*)(p);
    const float4 f1 = *(const float4*)(p + 4);
    const float4 f2 = *(const float4*)(p + 16);
    const float4 f3 = *(const float4*)(p + 20);
    a.e[0]  = (__bf16)f0.x; a.e[1]  = (__bf16)f0.y; a.e[2]  = (__bf16)f0.z; a.e[3]  = (__bf16)f0.w;
    a.e[4]  = (__bf16)f1.x; a.e[5]  = (__bf16)f1.y; a.e[6]  = (__bf16)f1.z; a.e[7]  = (__bf16)f1.w;
    a.e[8]  = (__bf16)f2.x; a.e[9]  = (__bf16)f2.y; a.e[10] = (__bf16)f2.z; a.e[11] = (__bf16)f2.w;
    a.e[12] = (__bf16)f3.x; a.e[13] = (__bf16)f3.y; a.e[14] = (__bf16)f3.z; a.e[15] = (__bf16)f3.w;
    return a.v;
}

// ---------------------------------------------------------------------------
// Kernel 2: scores[b,t] = sum_u tanh(combo[b,u] + (values[b,t,:]@W2)[u])*Vw[u] + Vb
//
// grid (B, T/256), block 256 = 8 waves. Block stages the full 128 KB of
// bf16 B-fragments into LDS once (2 blocks/WGP fit in 320 KB LDS).
// Each wave owns TWO 16-row t-tiles (A = 2x8 fragments in VGPRs) so every
// B fragment fetched from LDS feeds two independent wmmas -> LDS traffic
// matches the 64-bank bandwidth at full wmma issue rate.
// 16 N-tiles x 8 K-steps x 2 tiles = 256 v_wmma_f32_16x16x32_bf16 per wave.
// values are read from HBM exactly once.
// ---------------------------------------------------------------------------
__global__ void __launch_bounds__(256)
score_kernel(const float* __restrict__ values,
             const __bf16* __restrict__ w2pack,
             const float* __restrict__ combo,
             const float* __restrict__ Vw,
             const float* __restrict__ Vb,
             float* __restrict__ scores) {
    extern __shared__ __align__(16) char smem[];   // 131072 B of B-fragments
    __bf16* bsh = (__bf16*)smem;

    const int tid   = threadIdx.x;
    const int lane  = tid & 31;
    const int wave  = tid >> 5;
    const int b     = blockIdx.x;
    const int tbase = blockIdx.y * 256 + wave * 32;   // this wave's 32 rows
    const int khalf = (lane < 16) ? 0 : 8;

    // Cooperative copy of all B fragments global -> LDS (8192 x 16B).
    {
        uint4*       dst = (uint4*)smem;
        const uint4* src = (const uint4*)w2pack;
#pragma unroll
        for (int i = 0; i < 32; ++i)
            dst[tid + i * 256] = src[tid + i * 256];
    }

    // Load & convert this lane's A fragments for both t-tiles (128 VGPRs).
    const float* vrow0 = values + ((size_t)b * T_SZ + tbase + (lane & 15)) * D_SZ;
    const float* vrow1 = vrow0 + 16 * D_SZ;
    v16bf afrag0[8], afrag1[8];
#pragma unroll
    for (int ks = 0; ks < 8; ++ks) {
        afrag0[ks] = load_a_frag(vrow0 + ks * 32 + khalf);
        afrag1[ks] = load_a_frag(vrow1 + ks * 32 + khalf);
    }

    float srow0[8], srow1[8];
#pragma unroll
    for (int i = 0; i < 8; ++i) { srow0[i] = 0.0f; srow1[i] = 0.0f; }

    const float* combo_b = combo + b * U_SZ;

    __syncthreads();   // B fragments resident in LDS

    for (int nt = 0; nt < 16; ++nt) {
        v8f c0 = {}, c1 = {};
        const __bf16* bbase = bsh + ((nt * 8) * 32 + lane) * 16;
#pragma unroll
        for (int ks = 0; ks < 8; ++ks) {
            const v16bf bfrag = *(const v16bf*)(bbase + ks * 512);
            c0 = __builtin_amdgcn_wmma_f32_16x16x32_bf16(
                     false, afrag0[ks], false, bfrag, (short)0, c0, false, false);
            c1 = __builtin_amdgcn_wmma_f32_16x16x32_bf16(
                     false, afrag1[ks], false, bfrag, (short)0, c1, false, false);
        }
        const int   n   = nt * 16 + (lane & 15);
        const float cmb = combo_b[n];
        const float vw  = Vw[n];
#pragma unroll
        for (int i = 0; i < 8; ++i) {
            srow0[i] += fast_tanh(c0[i] + cmb) * vw;   // row M = i + (lane<16?0:8)
            srow1[i] += fast_tanh(c1[i] + cmb) * vw;
        }
    }

    // Reduce over the 16 lanes of each half-wave (same rows, different N).
#pragma unroll
    for (int i = 0; i < 8; ++i) {
        float v0 = srow0[i], v1 = srow1[i];
        v0 += __shfl_xor(v0, 1, 32); v1 += __shfl_xor(v1, 1, 32);
        v0 += __shfl_xor(v0, 2, 32); v1 += __shfl_xor(v1, 2, 32);
        v0 += __shfl_xor(v0, 4, 32); v1 += __shfl_xor(v1, 4, 32);
        v0 += __shfl_xor(v0, 8, 32); v1 += __shfl_xor(v1, 8, 32);
        srow0[i] = v0; srow1[i] = v1;
    }
    if ((lane & 15) == 0) {
        const int   roff = (lane >= 16) ? 8 : 0;   // lanes16-31 held rows 8..15
        const float vb   = Vb[0];
        float* sc = scores + b * T_SZ + tbase + roff;
#pragma unroll
        for (int i = 0; i < 8; ++i) {
            sc[i]      = srow0[i] + vb;
            sc[16 + i] = srow1[i] + vb;
        }
    }
}

// ---------------------------------------------------------------------------
// Kernel 3: softmax over T per batch; writes attention weights into d_out
// and zeroes the context region for kernel 4's atomics (graph-replay safe).
// ---------------------------------------------------------------------------
__global__ void softmax_kernel(const float* __restrict__ scores,
                               float* __restrict__ weights,  // d_out + B*D
                               float* __restrict__ ctx) {    // d_out
    __shared__ float red[256];
    const int b   = blockIdx.x;
    const int tid = threadIdx.x;

    ctx[b * D_SZ + tid] = 0.0f;

    float loc[8];
    float m = -INFINITY;
#pragma unroll
    for (int j = 0; j < 8; ++j) {
        loc[j] = scores[b * T_SZ + j * 256 + tid];
        m = fmaxf(m, loc[j]);
    }
    red[tid] = m;
    __syncthreads();
    for (int s = 128; s > 0; s >>= 1) {
        if (tid < s) red[tid] = fmaxf(red[tid], red[tid + s]);
        __syncthreads();
    }
    m = red[0];
    __syncthreads();

    float sum = 0.0f;
#pragma unroll
    for (int j = 0; j < 8; ++j) {
        loc[j] = __expf(loc[j] - m);
        sum += loc[j];
    }
    red[tid] = sum;
    __syncthreads();
    for (int s = 128; s > 0; s >>= 1) {
        if (tid < s) red[tid] += red[tid + s];
        __syncthreads();
    }
    const float inv = 1.0f / red[0];
#pragma unroll
    for (int j = 0; j < 8; ++j)
        weights[b * T_SZ + j * 256 + tid] = loc[j] * inv;
}

// ---------------------------------------------------------------------------
// Kernel 4: context[b,d] = sum_t w[b,t] * values[b,t,d]
// grid (B, 8): each block handles a 256-timestep slice, atomicAdd partials.
// Second pass over values is largely L2-resident (134 MB < 192 MB L2).
// ---------------------------------------------------------------------------
__global__ void context_kernel(const float* __restrict__ values,
                               const float* __restrict__ weights,
                               float* __restrict__ ctx) {
    __shared__ float wsh[256];
    const int b  = blockIdx.x;
    const int t0 = blockIdx.y * 256;
    const int d  = threadIdx.x;

    wsh[d] = weights[b * T_SZ + t0 + d];
    __syncthreads();

    const float* vp = values + ((size_t)b * T_SZ + t0) * D_SZ + d;
    float acc = 0.0f;
#pragma unroll 8
    for (int t = 0; t < 256; ++t)
        acc += wsh[t] * vp[(size_t)t * D_SZ];

    atomicAdd(&ctx[b * D_SZ + d], acc);
}

// ---------------------------------------------------------------------------
// Launch
// ---------------------------------------------------------------------------
extern "C" void kernel_launch(void* const* d_in, const int* in_sizes, int n_in,
                              void* d_out, int out_size, void* d_ws, size_t ws_size,
                              hipStream_t stream) {
    const float* query  = (const float*)d_in[0];  // [B, D]
    const float* values = (const float*)d_in[1];  // [B, T, D]
    const float* W1w = (const float*)d_in[2];     // [D, U]
    const float* W1b = (const float*)d_in[3];     // [U]
    const float* W2w = (const float*)d_in[4];     // [D, U]
    const float* W2b = (const float*)d_in[5];     // [U]
    const float* Vw  = (const float*)d_in[6];     // [U, 1]
    const float* Vb  = (const float*)d_in[7];     // [1]

    float* out_ctx = (float*)d_out;                     // [B, D]
    float* out_w   = (float*)d_out + B_SZ * D_SZ;       // [B, T, 1]

    // Workspace layout
    char* ws = (char*)d_ws;
    float*  combo  = (float*)(ws);                          // 64 KB
    __bf16* w2pack = (__bf16*)(ws + 65536);                 // 128 KB
    float*  scores = (float*)(ws + 65536 + 131072);         // 512 KB

    combo_kernel<<<B_SZ, U_SZ, 0, stream>>>(query, W1w, W1b, W2b, combo);
    pack_w2_kernel<<<128, 32, 0, stream>>>(W2w, w2pack);

    score_kernel<<<dim3(B_SZ, T_SZ / 256), 256, 131072, stream>>>(
        values, w2pack, combo, Vw, Vb, scores);

    softmax_kernel<<<B_SZ, 256, 0, stream>>>(scores, out_w, out_ctx);

    context_kernel<<<dim3(B_SZ, 8), 256, 0, stream>>>(values, out_w, out_ctx);
}
